// GCN_240518168947
// MI455X (gfx1250) — matrix operand order repaired and checked by reading, not compiled, not run on hardware
//
#include <hip/hip_runtime.h>
#include <math.h>

typedef float     v2f  __attribute__((ext_vector_type(2)));
typedef float     v8f  __attribute__((ext_vector_type(8)));
typedef _Float16  v16h __attribute__((ext_vector_type(16)));

#define N_NODES 50000
#define N_EDGES 1600000

#if __has_builtin(__builtin_amdgcn_wmma_f32_16x16x4_f32)
#define HAVE_F32_WMMA 1
#endif

// ---------------------------------------------------------------------------
// agg[n*C + c] = b[c]   (bias-initialized accumulator for scatter-add)
// ---------------------------------------------------------------------------
__global__ void bias_init(float* __restrict__ agg, const float* __restrict__ b,
                          int C, int total) {
  int gid = blockIdx.x * blockDim.x + threadIdx.x;
  if (gid < total) agg[gid] = b[gid % C];
}

// ---------------------------------------------------------------------------
// T[n, o] = sum_k act(X[n,k]) * W[o,k]     (act = ReLU if RELU)
// X: [N,64] row-major, W: [Cout,64] row-major, T: [N, outStride]
// Block: 128 threads = 4 waves. Block handles 16 node rows; wave w handles
// output columns [16w, 16w+16). Uses V_WMMA_F32_16X16X4_F32 (fp32 matrix
// pipe) — 16 WMMA ops per tile for K=64.
// ---------------------------------------------------------------------------
template <bool RELU>
__global__ __launch_bounds__(128) void linear_wmma(
    const float* __restrict__ X, const float* __restrict__ W,
    float* __restrict__ T, int Cout, int outStride) {
  __shared__ float Xs[16][68];   // pad 68: stride mod 64 == 4 -> conflict-free
  __shared__ float Ws[64][68];

  const int tid = threadIdx.x;
  const int rowBase = blockIdx.x * 16;

  // Cooperative, coalesced staging of the 16x64 X tile.
  for (int idx = tid; idx < 16 * 64; idx += 128) {
    int r = idx >> 6, c = idx & 63;
    float v = X[(size_t)(rowBase + r) * 64 + c];
    if (RELU) v = fmaxf(v, 0.0f);
    Xs[r][c] = v;
  }
  // Stage W (zero-fill rows >= Cout so layer-3's 7x64 W works unchanged).
  for (int idx = tid; idx < 64 * 64; idx += 128) {
    int r = idx >> 6, c = idx & 63;
    Ws[r][c] = (r < Cout) ? W[(size_t)r * 64 + c] : 0.0f;
  }
  __syncthreads();

  const int wave  = tid >> 5;
  const int lane  = tid & 31;
  const int half  = lane >> 4;   // lane group 0/1
  const int l15   = lane & 15;   // M index for A-frag, N index for B/D-frag
  const int nTiles = (Cout + 15) >> 4;
  if (wave >= nTiles) return;
  const int colBase = wave * 16;

  v8f acc = {};
#ifdef HAVE_F32_WMMA
  // fp32 path: A 16x4 (2 VGPR), B 4x16 (2 VGPR), K advances by 4.
  for (int k0 = 0; k0 < 64; k0 += 4) {
    const int kk = k0 + 2 * half;
    v2f a, b;
    a.x = Xs[l15][kk];
    a.y = Xs[l15][kk + 1];
    b.x = Ws[colBase + l15][kk];
    b.y = Ws[colBase + l15][kk + 1];
    acc = __builtin_amdgcn_wmma_f32_16x16x4_f32(
        false, a, false, b, (short)0, acc, false, false);
  }
#else
  // f16 fallback: two 16x16x32 WMMAs cover K=64 (fp32 accumulate).
  for (int k0 = 0; k0 < 64; k0 += 32) {
    v16h a, b;
    for (int j = 0; j < 8; ++j) {
      a[j]     = (_Float16)Xs[l15][k0 + 8 * half + j];
      a[j + 8] = (_Float16)Xs[l15][k0 + 16 + 8 * half + j];
    }
    for (int j = 0; j < 8; ++j) {
      b[j]     = (_Float16)Ws[colBase + l15][k0 + 8 * half + j];
      b[j + 8] = (_Float16)Ws[colBase + l15][k0 + 16 + 8 * half + j];
    }
    acc = __builtin_amdgcn_wmma_f32_16x16x32_f16(
        false, a, false, b, (short)0, acc, false, false);
  }
#endif

  // D layout: lanes 0-15 -> M = i, lanes 16-31 -> M = 8 + i ; N = lane&15.
  const int col = colBase + l15;
  if (col < Cout) {
    for (int i = 0; i < 8; ++i)
      T[(size_t)(rowBase + 8 * half + i) * outStride + col] = acc[i];
  }
}

// ---------------------------------------------------------------------------
// Scatter-add for C=64: 16 threads per edge, float4 per thread (coalesced
// 256B gather per edge, fp32 atomics land in L2).
// ---------------------------------------------------------------------------
__global__ void scatter64(const float* __restrict__ T,
                          const int* __restrict__ src,
                          const int* __restrict__ dst,
                          float* __restrict__ agg) {
  int gid = blockIdx.x * blockDim.x + threadIdx.x;
  int e = gid >> 4;
  if (e >= N_EDGES) return;
  int c = (gid & 15) * 4;
  int s = src[e], d = dst[e];
  const float4 v = *(const float4*)(T + (size_t)s * 64 + c);
  float* out = agg + (size_t)d * 64 + c;
  atomicAdd(out + 0, v.x);
  atomicAdd(out + 1, v.y);
  atomicAdd(out + 2, v.z);
  atomicAdd(out + 3, v.w);
}

// Scatter-add for C=7: 8 threads per edge, lanes with c<7 active.
__global__ void scatter7(const float* __restrict__ T,
                         const int* __restrict__ src,
                         const int* __restrict__ dst,
                         float* __restrict__ agg) {
  int gid = blockIdx.x * blockDim.x + threadIdx.x;
  int e = gid >> 3;
  if (e >= N_EDGES) return;
  int c = gid & 7;
  if (c >= 7) return;
  int s = src[e], d = dst[e];
  atomicAdd(agg + (size_t)d * 7 + c, T[(size_t)s * 7 + c]);
}

// ---------------------------------------------------------------------------
// Row-wise log_softmax over 7 logits.
// ---------------------------------------------------------------------------
__global__ void logsoftmax7(const float* __restrict__ logits,
                            float* __restrict__ out) {
  int n = blockIdx.x * blockDim.x + threadIdx.x;
  if (n >= N_NODES) return;
  float v[7];
  float m = -INFINITY;
  for (int i = 0; i < 7; ++i) {
    v[i] = logits[(size_t)n * 7 + i];
    m = fmaxf(m, v[i]);
  }
  float s = 0.0f;
  for (int i = 0; i < 7; ++i) s += __expf(v[i] - m);
  float l = __logf(s);
  for (int i = 0; i < 7; ++i) out[(size_t)n * 7 + i] = v[i] - m - l;
}

// ---------------------------------------------------------------------------
extern "C" void kernel_launch(void* const* d_in, const int* in_sizes, int n_in,
                              void* d_out, int out_size, void* d_ws,
                              size_t ws_size, hipStream_t stream) {
  (void)in_sizes; (void)n_in; (void)out_size; (void)ws_size;
  const float* x  = (const float*)d_in[0];
  const int*   ei = (const int*)d_in[1];
  const float* W1 = (const float*)d_in[2];
  const float* b1 = (const float*)d_in[3];
  const float* W2 = (const float*)d_in[4];
  const float* b2 = (const float*)d_in[5];
  const float* W3 = (const float*)d_in[6];
  const float* b3 = (const float*)d_in[7];
  float* out = (float*)d_out;
  const int* src = ei;
  const int* dst = ei + N_EDGES;

  float* ws   = (float*)d_ws;
  float* t    = ws;                                // [50000,64] (reused, stride 7 in L3)
  float* agg1 = t    + (size_t)N_NODES * 64;       // [50000,64]
  float* agg2 = agg1 + (size_t)N_NODES * 64;       // [50000,64]
  float* agg3 = agg2 + (size_t)N_NODES * 64;       // [50000,7]

  const int nodeTiles = N_NODES / 16;              // 3125 (exact)

  // --- Layer 1 ---
  bias_init<<<(N_NODES * 64 + 255) / 256, 256, 0, stream>>>(agg1, b1, 64, N_NODES * 64);
  linear_wmma<false><<<nodeTiles, 128, 0, stream>>>(x, W1, t, 64, 64);
  scatter64<<<(N_EDGES * 16 + 255) / 256, 256, 0, stream>>>(t, src, dst, agg1);

  // --- Layer 2 (ReLU folded into the A-matrix load) ---
  bias_init<<<(N_NODES * 64 + 255) / 256, 256, 0, stream>>>(agg2, b2, 64, N_NODES * 64);
  linear_wmma<true><<<nodeTiles, 128, 0, stream>>>(agg1, W2, t, 64, 64);
  scatter64<<<(N_EDGES * 16 + 255) / 256, 256, 0, stream>>>(t, src, dst, agg2);

  // --- Layer 3 (Cout = 7) ---
  bias_init<<<(N_NODES * 7 + 255) / 256, 256, 0, stream>>>(agg3, b3, 7, N_NODES * 7);
  linear_wmma<true><<<nodeTiles, 128, 0, stream>>>(agg2, W3, t, 7, 7);
  scatter7<<<(N_EDGES * 8 + 255) / 256, 256, 0, stream>>>(t, src, dst, agg3);

  // --- log_softmax ---
  logsoftmax7<<<(N_NODES + 255) / 256, 256, 0, stream>>>(agg3, out);
}
